// LSTM_90409061581270
// MI455X (gfx1250) — compile-verified, matrix operand order checked by reading, not run on hardware
//
#include <hip/hip_runtime.h>
#include <hip/hip_bf16.h>

#define B_SZ 512
#define T_SZ 256
#define EMB  300
#define KPAD 320          // EMB padded to multiple of 32 (zero fill)
#define HID  128
#define G4   512          // 4*HID
#define NCLS 2
#define MT   128          // proj kernel M tile
#define NT   64           // proj kernel N tile

typedef __attribute__((ext_vector_type(8)))  float  v8f;
typedef __attribute__((ext_vector_type(16))) __bf16 v16bf;

union alignas(32) Frag16 { v16bf bf; unsigned short us[16]; uint4 q[2]; };

__device__ __forceinline__ unsigned short f2bf(float f) {
  unsigned u = __float_as_uint(f);
  unsigned r = u + 0x7FFFu + ((u >> 16) & 1u);   // round-to-nearest-even
  return (unsigned short)(r >> 16);
}
__device__ __forceinline__ float bf2f(unsigned short h) {
  return __uint_as_float(((unsigned)h) << 16);
}
__device__ __forceinline__ v8f vzero8() {
  v8f v;
#pragma unroll
  for (int i = 0; i < 8; ++i) v[i] = 0.f;
  return v;
}
__device__ __forceinline__ float fsigmoid(float x) {
  return 1.f / (1.f + __expf(-x));
}
__device__ __forceinline__ float ftanh(float x) {
  float e = __expf(2.f * x);
  return 1.f - 2.f / (e + 1.f);
}

// Load a 16x32 bf16 WMMA A/B fragment from a row-major ushort buffer.
// ISA layout: lane L holds row (L&15); lanes 0-15 carry K=[0..7]+[16..23],
// lanes 16-31 carry K=[8..15]+[24..31]; 2 bf16 per VGPR.
// Both 8-element K runs are 16B-contiguous -> two 128-bit loads.
__device__ __forceinline__ v16bf load_frag(const unsigned short* base, int row,
                                           int stride, int k0, int lane) {
  Frag16 f;
  int kb = (lane >> 4) << 3;                       // 0 or 8
  const uint4* q = (const uint4*)(base + row * stride + k0 + kb);
  f.q[0] = q[0];                                   // K = kb+0 .. kb+7
  f.q[1] = q[2];                                   // K = kb+16 .. kb+23
  return f.bf;
}

// ---------------- kernel 0: weight conversion ----------------
__global__ void prep_kernel(const float* __restrict__ W_ih, const float* __restrict__ W_hh,
                            const float* __restrict__ b_ih, const float* __restrict__ b_hh,
                            unsigned short* __restrict__ w_ih_bf,
                            unsigned short* __restrict__ w_hh_bf,
                            float* __restrict__ bias) {
  int i = blockIdx.x * blockDim.x + threadIdx.x;
  int stride = gridDim.x * blockDim.x;
  for (int e = i; e < G4 * KPAD; e += stride) {
    int n = e / KPAD, k = e - n * KPAD;
    w_ih_bf[e] = (k < EMB) ? f2bf(W_ih[n * EMB + k]) : (unsigned short)0;
  }
  for (int e = i; e < G4 * HID; e += stride) w_hh_bf[e] = f2bf(W_hh[e]);
  for (int e = i; e < G4; e += stride) bias[e] = b_ih[e] + b_hh[e];
}

// ---------------- kernel 1: embedding gather + input projection GEMM --------
// x_proj[m][n], m = t*512 + b, n = gate index. Output in bf16.
__launch_bounds__(256)
__global__ void proj_kernel(const int* __restrict__ texts, const float* __restrict__ emb,
                            const unsigned short* __restrict__ w_ih_bf,
                            const float* __restrict__ bias,
                            unsigned short* __restrict__ x_proj) {
  __shared__ alignas(16) unsigned short As[MT * KPAD];  // 80 KB
  __shared__ alignas(16) unsigned short Bs[NT * KPAD];  // 40 KB
  const int m0 = blockIdx.x * MT;
  const int n0 = blockIdx.y * NT;
  const int tid = threadIdx.x;

  // stage B tile (contiguous rows of w_ih_bf)
  {
    const uint4* src = (const uint4*)(w_ih_bf + n0 * KPAD);
    uint4* dst = (uint4*)Bs;
#pragma unroll 2
    for (int v = tid; v < NT * KPAD / 8; v += 256) dst[v] = src[v];
  }
  // stage A tile: gather embedding rows, convert to bf16, zero-pad K
  {
    const int t = m0 >> 9;                    // MT=128 divides 512: one t per tile
    for (int e = tid; e < MT * KPAD; e += 256) {
      int r = e / KPAD, c = e - r * KPAD;
      int b = (m0 + r) & (B_SZ - 1);
      int tok = texts[b * T_SZ + t];
      As[e] = (c < EMB) ? f2bf(emb[tok * EMB + c]) : (unsigned short)0;
    }
  }
  __syncthreads();

  const int wave = tid >> 5, lane = tid & 31;
  const int rsel = lane & 15;
  v8f acc[4] = {vzero8(), vzero8(), vzero8(), vzero8()};

#pragma unroll
  for (int ks = 0; ks < KPAD / 32; ++ks) {
    v16bf a = load_frag(As, (wave << 4) + rsel, KPAD, ks * 32, lane);
#pragma unroll
    for (int g = 0; g < 4; ++g) {
      v16bf b = load_frag(Bs, (g << 4) + rsel, KPAD, ks * 32, lane);
      acc[g] = __builtin_amdgcn_wmma_f32_16x16x32_bf16(false, a, false, b,
                                                       (short)0, acc[g], false, false);
    }
  }

  const int hi = lane >> 4;
#pragma unroll
  for (int g = 0; g < 4; ++g) {
    int ng = n0 + (g << 4) + rsel;
    float bv = bias[ng];
#pragma unroll
    for (int r = 0; r < 8; ++r) {
      int m = m0 + (wave << 4) + r + (hi << 3);
      x_proj[(size_t)m * G4 + ng] = f2bf(acc[g][r] + bv);
    }
  }
}

// ---------------- kernel 2: recurrent LSTM scan ----------------
// 32 WGs x 8 waves; WG owns 16 batch rows for all 256 steps. Wave w computes
// gate tiles {w, w+8, w+16, w+24} = (i,f,g,o) for h-columns 16w..16w+15, so
// the cell update is register-resident (c state never leaves VGPRs).
__launch_bounds__(256)
__global__ void lstm_scan_kernel(const int* __restrict__ lengths,
                                 const unsigned short* __restrict__ w_hh_bf,
                                 const unsigned short* __restrict__ x_proj,
                                 float* __restrict__ hT) {
  __shared__ alignas(16) unsigned short Whh[G4 * HID];  // 128 KB, resident
  __shared__ alignas(16) unsigned short Hs[16 * HID];   // 4 KB h exchange
  const int tid = threadIdx.x, wave = tid >> 5, lane = tid & 31;
  const int b0 = blockIdx.x << 4;
  const int hi = lane >> 4, ncol = lane & 15;

  {
    const uint4* src = (const uint4*)w_hh_bf;
    uint4* dst = (uint4*)Whh;
#pragma unroll 4
    for (int v = tid; v < G4 * HID / 8; v += 256) dst[v] = src[v];
  }
  for (int e = tid; e < 16 * HID; e += 256) Hs[e] = 0;

  int lenr[8];
#pragma unroll
  for (int r = 0; r < 8; ++r) lenr[r] = lengths[b0 + r + (hi << 3)];

  v8f c = vzero8(), h = vzero8();
  __syncthreads();

  for (int t = 0; t < T_SZ; ++t) {
    // accumulator init = precomputed input projection (bf16 -> f32)
    v8f acc[4];
    const unsigned short* xp = x_proj + ((size_t)t * B_SZ + b0) * G4;
#pragma unroll
    for (int g = 0; g < 4; ++g) {
      int ng = ((wave + (g << 3)) << 4) + ncol;
#pragma unroll
      for (int r = 0; r < 8; ++r)
        acc[g][r] = bf2f(xp[(size_t)(r + (hi << 3)) * G4 + ng]);
    }
    // prefetch next step's 16KB x_proj slab (global_prefetch_b8)
    if (t + 1 < T_SZ)
      __builtin_prefetch((const char*)(xp + (size_t)B_SZ * G4) + tid * 64, 0, 0);

    // gates += h_prev @ W_hh^T  (K = 128 -> 4 k-steps)
#pragma unroll
    for (int ks = 0; ks < 4; ++ks) {
      v16bf a = load_frag(Hs, ncol, HID, ks * 32, lane);
#pragma unroll
      for (int g = 0; g < 4; ++g) {
        v16bf b = load_frag(Whh, ((wave + (g << 3)) << 4) + ncol, HID, ks * 32, lane);
        acc[g] = __builtin_amdgcn_wmma_f32_16x16x32_bf16(false, a, false, b,
                                                         (short)0, acc[g], false, false);
      }
    }

    // cell update, packed-sequence masking (freeze state once t >= length)
#pragma unroll
    for (int r = 0; r < 8; ++r) {
      float iv = fsigmoid(acc[0][r]);
      float fv = fsigmoid(acc[1][r]);
      float gv = ftanh(acc[2][r]);
      float ov = fsigmoid(acc[3][r]);
      float cn = fv * c[r] + iv * gv;
      float hn = ov * ftanh(cn);
      bool valid = t < lenr[r];
      c[r] = valid ? cn : c[r];
      h[r] = valid ? hn : h[r];
    }

    __syncthreads();                         // all waves done reading Hs
#pragma unroll
    for (int r = 0; r < 8; ++r)
      Hs[(r + (hi << 3)) * HID + (wave << 4) + ncol] = f2bf(h[r]);
    __syncthreads();                         // Hs ready for next step
  }

#pragma unroll
  for (int r = 0; r < 8; ++r)
    hT[(size_t)(b0 + r + (hi << 3)) * HID + (wave << 4) + ncol] = h[r];
}

// ---------------- kernel 3: final classifier ----------------
__global__ void fc_kernel(const float* __restrict__ hT, const float* __restrict__ W_fc,
                          const float* __restrict__ b_fc, float* __restrict__ out) {
  int i = blockIdx.x * blockDim.x + threadIdx.x;
  if (i >= B_SZ * NCLS) return;
  int b = i >> 1, c = i & 1;
  float s = b_fc[c];
#pragma unroll 4
  for (int j = 0; j < HID; ++j) s += hT[b * HID + j] * W_fc[c * HID + j];
  out[i] = s;
}

extern "C" void kernel_launch(void* const* d_in, const int* in_sizes, int n_in,
                              void* d_out, int out_size, void* d_ws, size_t ws_size,
                              hipStream_t stream) {
  const int*   texts   = (const int*)d_in[0];
  const int*   lengths = (const int*)d_in[1];
  const float* emb     = (const float*)d_in[2];
  const float* W_ih    = (const float*)d_in[3];
  const float* W_hh    = (const float*)d_in[4];
  const float* b_ih    = (const float*)d_in[5];
  const float* b_hh    = (const float*)d_in[6];
  const float* W_fc    = (const float*)d_in[7];
  const float* b_fc    = (const float*)d_in[8];

  char* ws = (char*)d_ws;
  unsigned short* w_ih_bf = (unsigned short*)ws;                       // 327,680 B
  unsigned short* w_hh_bf = (unsigned short*)(ws + 327680);            // 131,072 B
  float*          bias    = (float*)(ws + 458752);                     //   2,048 B
  unsigned short* x_proj  = (unsigned short*)(ws + 460800);            // 134,217,728 B
  float*          hT      = (float*)(ws + 460800 + 134217728);         //     262,144 B

  prep_kernel<<<128, 256, 0, stream>>>(W_ih, W_hh, b_ih, b_hh, w_ih_bf, w_hh_bf, bias);
  dim3 g1(B_SZ * T_SZ / MT, G4 / NT);   // (1024, 8)
  proj_kernel<<<g1, 256, 0, stream>>>(texts, emb, w_ih_bf, bias, x_proj);
  lstm_scan_kernel<<<B_SZ / 16, 256, 0, stream>>>(lengths, w_hh_bf, x_proj, hT);
  fc_kernel<<<(B_SZ * NCLS + 255) / 256, 256, 0, stream>>>(hT, W_fc, b_fc, (float*)d_out);
}
